// BatchHardTripletLoss_19636590478149
// MI455X (gfx1250) — matrix operand order repaired and checked by reading, not compiled, not run on hardware
//
#include <hip/hip_runtime.h>
#include <hip/hip_bf16.h>

// ---------------------------------------------------------------------------
// Batch-hard triplet loss, fused Gram-GEMM via CDNA5 WMMA (gfx1250, wave32).
// gram = e@e.T computed with 3x-BF16 split (hi*hi + hi*lo + lo*hi) using
// v_wmma_f32_16x16x32_bf16; distances + hardest-pos/neg fused in-register.
// ---------------------------------------------------------------------------

#define BATCH 4096
#define DIM   512
#define MARGIN 0.5f
#define EPSV   1e-6f
#define BIGV   1e9f

typedef __bf16 v16bf __attribute__((ext_vector_type(16)));
typedef float  v8f   __attribute__((ext_vector_type(8)));

union Frag32 {              // 32-byte WMMA operand fragment (16 x bf16)
    uint4 q[2];
    v16bf v;
};

// ---------------------------------------------------------------------------
// Kernel 1: per-row sum / sum-of-squares (fp32 exact) + bf16 hi/lo split.
// One wave32 per row; 8 rows per 256-thread block.
// ---------------------------------------------------------------------------
__global__ __launch_bounds__(256)
void triplet_prep_kernel(const float* __restrict__ emb,
                         __bf16* __restrict__ ehi, __bf16* __restrict__ elo,
                         float* __restrict__ sq, float* __restrict__ s)
{
    const int w    = threadIdx.x >> 5;
    const int lane = threadIdx.x & 31;
    const int row  = blockIdx.x * 8 + w;
    const size_t base = (size_t)row * DIM;

    float acc2 = 0.0f, acc1 = 0.0f;
#pragma unroll
    for (int j = 0; j < DIM / 32; ++j) {
        const int idx = j * 32 + lane;
        const float x = emb[base + idx];
        acc2 += x * x;
        acc1 += x;
        const __bf16 hi = (__bf16)x;
        const __bf16 lo = (__bf16)(x - (float)hi);
        ehi[base + idx] = hi;
        elo[base + idx] = lo;
    }
#pragma unroll
    for (int m = 16; m >= 1; m >>= 1) {
        acc2 += __shfl_xor(acc2, m, 32);
        acc1 += __shfl_xor(acc1, m, 32);
    }
    if (lane == 0) { sq[row] = acc2; s[row] = acc1; }
}

// ---------------------------------------------------------------------------
// Kernel 2: fused Gram GEMM + masked hardest-pos / hardest-neg reductions.
// One block per 16-row M-tile. A-tile (hi+lo) staged in LDS, shared by 8
// waves; each wave sweeps 32 N-tiles of 16 columns with WMMA bf16.
// ---------------------------------------------------------------------------
__global__ __launch_bounds__(256)
void triplet_gram_hard_kernel(const __bf16* __restrict__ ehi,
                              const __bf16* __restrict__ elo,
                              const float* __restrict__ sq,
                              const float* __restrict__ s,
                              const int*  __restrict__ labels,
                              float* __restrict__ gHp,
                              float* __restrict__ gHn)
{
    __shared__ alignas(16) __bf16 sAhi[16 * DIM];   // 16 KB
    __shared__ alignas(16) __bf16 sAlo[16 * DIM];   // 16 KB
    __shared__ float redHp[8][16];
    __shared__ float redHn[8][16];

    const int tid  = threadIdx.x;
    const int w    = tid >> 5;           // wave id 0..7
    const int lane = tid & 31;
    const int half = lane >> 4;          // 0 or 1
    const int lx   = lane & 15;
    const int mTile = blockIdx.x;
    const size_t mBase = (size_t)mTile * 16 * DIM;

    // --- cooperative A-tile stage (16 rows, hi + lo) into LDS ---
    {
        const uint2* gh = (const uint2*)(ehi + mBase);
        const uint2* gl = (const uint2*)(elo + mBase);
        uint2* sh = (uint2*)sAhi;
        uint2* sl = (uint2*)sAlo;
#pragma unroll
        for (int i = 0; i < 8; ++i) {    // 2048 uint2 per array / 256 threads
            const int li = tid + i * 256;
            sh[li] = gh[li];
            sl[li] = gl[li];
        }
    }
    __syncthreads();

    // Per-lane anchor-row stats: C/D element r maps to M = r + 8*half.
    float msq[8], msum[8];
    int   mlab[8], mrow[8];
#pragma unroll
    for (int r = 0; r < 8; ++r) {
        const int m = mTile * 16 + r + half * 8;
        mrow[r] = m; msq[r] = sq[m]; msum[r] = s[m]; mlab[r] = labels[m];
    }

    float hp[8], hn[8];
#pragma unroll
    for (int r = 0; r < 8; ++r) { hp[r] = -BIGV; hn[r] = BIGV; }

    const float epsC = (float)DIM * EPSV * EPSV;

    for (int t = 0; t < 32; ++t) {
        const int nTile = t * 8 + w;
        const int n     = nTile * 16 + lx;    // this lane's N column
        // B fragment: lane holds K = kb*32 + half*8 + (0..7) and +16.
        const __bf16* bh = ehi + (size_t)n * DIM + half * 8;
        const __bf16* bl = elo + (size_t)n * DIM + half * 8;

        if (t + 1 < 32) {                      // warm next N-tile (L2->L0)
            const int nn = (t + 1) * 8 + w;
            __builtin_prefetch(ehi + (size_t)(nn * 16 + lx) * DIM, 0, 0);
            __builtin_prefetch(elo + (size_t)(nn * 16 + lx) * DIM, 0, 0);
        }

        v8f acc = {0.f, 0.f, 0.f, 0.f, 0.f, 0.f, 0.f, 0.f};

#pragma unroll
        for (int kb = 0; kb < DIM / 32; ++kb) {
            Frag32 ah, al, bhf, blf;
            // A fragment: row M = lx; K = kb*32 + half*8 + (0..7) and +16.
            const __bf16* ar = sAhi + (size_t)lx * DIM + kb * 32 + half * 8;
            const __bf16* al_p = sAlo + (size_t)lx * DIM + kb * 32 + half * 8;
            ah.q[0] = *(const uint4*)(ar);
            ah.q[1] = *(const uint4*)(ar + 16);
            al.q[0] = *(const uint4*)(al_p);
            al.q[1] = *(const uint4*)(al_p + 16);
            bhf.q[0] = *(const uint4*)(bh + kb * 32);
            bhf.q[1] = *(const uint4*)(bh + kb * 32 + 16);
            blf.q[0] = *(const uint4*)(bl + kb * 32);
            blf.q[1] = *(const uint4*)(bl + kb * 32 + 16);

            // 3x-BF16 fp32 emulation: hi*hi + hi*lo + lo*hi
            acc = __builtin_amdgcn_wmma_f32_16x16x32_bf16(
                false, ah.v, false, bhf.v, (short)0, acc, false, false);
            acc = __builtin_amdgcn_wmma_f32_16x16x32_bf16(
                false, ah.v, false, blf.v, (short)0, acc, false, false);
            acc = __builtin_amdgcn_wmma_f32_16x16x32_bf16(
                false, al.v, false, bhf.v, (short)0, acc, false, false);
        }

        // Convert Gram tile -> distances, apply masks, update running hp/hn.
        const float sqn = sq[n];
        const float sn  = s[n];
        const int   ln  = labels[n];
#pragma unroll
        for (int r = 0; r < 8; ++r) {
            float d2 = msq[r] + sqn - 2.0f * acc[r]
                     + 2.0f * EPSV * (msum[r] - sn) + epsC;
            const float d = sqrtf(fmaxf(d2, 0.0f));
            const bool same = (mlab[r] == ln);
            const bool diag = (mrow[r] == n);
            if (same && !diag) hp[r] = fmaxf(hp[r], d);
            if (!same)         hn[r] = fminf(hn[r], d);
        }
    }

    // Reduce across the 16 lanes (N positions) within each half.
#pragma unroll
    for (int r = 0; r < 8; ++r) {
#pragma unroll
        for (int mk = 8; mk >= 1; mk >>= 1) {
            hp[r] = fmaxf(hp[r], __shfl_xor(hp[r], mk, 32));
            hn[r] = fminf(hn[r], __shfl_xor(hn[r], mk, 32));
        }
    }
    if (lane == 0) {
#pragma unroll
        for (int r = 0; r < 8; ++r) { redHp[w][r] = hp[r]; redHn[w][r] = hn[r]; }
    }
    if (lane == 16) {
#pragma unroll
        for (int r = 0; r < 8; ++r) { redHp[w][8 + r] = hp[r]; redHn[w][8 + r] = hn[r]; }
    }
    __syncthreads();

    if (tid < 16) {
        float a = -BIGV, b = BIGV;
#pragma unroll
        for (int ww = 0; ww < 8; ++ww) {
            a = fmaxf(a, redHp[ww][tid]);
            b = fminf(b, redHn[ww][tid]);
        }
        gHp[mTile * 16 + tid] = a;
        gHn[mTile * 16 + tid] = b;
    }
}

// ---------------------------------------------------------------------------
// Kernel 3: final scalar: mean over valid anchors of relu(hp - hn + margin).
// ---------------------------------------------------------------------------
__global__ __launch_bounds__(256)
void triplet_finalize_kernel(const float* __restrict__ gHp,
                             const float* __restrict__ gHn,
                             float* __restrict__ out)
{
    __shared__ float sSum[256];
    __shared__ float sCnt[256];
    const int tid = threadIdx.x;
    float sum = 0.0f, cnt = 0.0f;
    for (int i = tid; i < BATCH; i += 256) {
        const float hp = gHp[i];
        const float hn = gHn[i];
        const bool valid = (hp > -0.5f * BIGV) && (hn < 0.5f * BIGV);
        if (valid) {
            const float l = hp - hn + MARGIN;
            sum += (l > 0.0f) ? l : 0.0f;
            cnt += 1.0f;
        }
    }
    sSum[tid] = sum; sCnt[tid] = cnt;
    __syncthreads();
    for (int st = 128; st > 0; st >>= 1) {
        if (tid < st) { sSum[tid] += sSum[tid + st]; sCnt[tid] += sCnt[tid + st]; }
        __syncthreads();
    }
    if (tid == 0) out[0] = (sCnt[0] > 0.0f) ? (sSum[0] / sCnt[0]) : 0.0f;
}

// ---------------------------------------------------------------------------
extern "C" void kernel_launch(void* const* d_in, const int* in_sizes, int n_in,
                              void* d_out, int out_size, void* d_ws, size_t ws_size,
                              hipStream_t stream) {
    (void)in_sizes; (void)n_in; (void)out_size; (void)ws_size;

    const float* emb    = (const float*)d_in[0];
    const int*   labels = (const int*)d_in[1];
    float*       out    = (float*)d_out;

    char* ws = (char*)d_ws;
    const size_t szE = (size_t)BATCH * DIM * sizeof(__bf16);   // 4 MB
    __bf16* ehi = (__bf16*)(ws);
    __bf16* elo = (__bf16*)(ws + szE);
    float*  sq  = (float*)(ws + 2 * szE);
    float*  s   = (float*)(ws + 2 * szE + BATCH * sizeof(float));
    float*  gHp = (float*)(ws + 2 * szE + 2 * BATCH * sizeof(float));
    float*  gHn = (float*)(ws + 2 * szE + 3 * BATCH * sizeof(float));

    triplet_prep_kernel<<<BATCH / 8, 256, 0, stream>>>(emb, ehi, elo, sq, s);
    triplet_gram_hard_kernel<<<BATCH / 16, 256, 0, stream>>>(ehi, elo, sq, s,
                                                             labels, gHp, gHn);
    triplet_finalize_kernel<<<1, 256, 0, stream>>>(gHp, gHn, out);
}
